// MambaRNN_14096082665952
// MI455X (gfx1250) — matrix operand order repaired and checked by reading, not compiled, not run on hardware
//
#include <hip/hip_runtime.h>
#include <hip/hip_bf16.h>

// ---------------- model dims (match reference) ----------------
#define V_   50280
#define D_   768
#define DI_  1536
#define N_   16
#define KC_  4
#define R_   48
#define NL_  24
#define B_   2
#define L_   256
#define T_   (B_ * L_)
#define XDBW (R_ + 2 * N_)   // 80
#define EPS_ 1e-5f

typedef __bf16 bf16;
typedef __attribute__((ext_vector_type(16))) bf16  v16bf;
typedef __attribute__((ext_vector_type(8)))  bf16  v8bf;
typedef __attribute__((ext_vector_type(8)))  float v8f;
typedef __attribute__((ext_vector_type(4)))  unsigned int u32x4;
typedef __attribute__((ext_vector_type(8)))  unsigned int u32x8;

union AF { v16bf v; v8bf h[2]; };

__device__ __forceinline__ v8bf bzero8() {
  v8bf z;
#pragma unroll
  for (int i = 0; i < 8; ++i) z[i] = (bf16)0.0f;
  return z;
}

__device__ __forceinline__ v8bf ld8(const bf16* p, bool ok) {
  return ok ? *(const v8bf*)p : bzero8();
}

__device__ __forceinline__ float siluf(float x) { return x / (1.0f + __expf(-x)); }

// ---------------- WMMA GEMM: C[MxN] = A[MxK](bf16) * Bt[NxK](bf16, pre-transposed) --------
// All fragments are contiguous global loads: no LDS, no barriers.
// Block: 256 threads = 8 waves as 4(row) x 2(col); wave tile 32x64 = 2x4 WMMA tiles.
// Block tile: 128 x 128. M must be a multiple of 128 (M = 512 here).
// EPI: 0 = none, 1 = out = aux[idx] + acc (residual), 2 = out = softplus(acc + aux[col])
#define BMX 128
#define BNX 128
#define BKX 32

template <int EPI>
__global__ __launch_bounds__(256) void gemm_bf16_wmma(
    const bf16* __restrict__ A, const bf16* __restrict__ Bt,
    float* __restrict__ C, const float* __restrict__ aux,
    int Ncols, int Kdim, int lda, int ldbt, int ldc) {
  const int lane = threadIdx.x & 31;
  const int wave = threadIdx.x >> 5;
  const int wrow = wave >> 1;                 // 0..3
  const int wcol = wave & 1;                  // 0..1
  const int m0   = blockIdx.y * BMX + wrow * 32;
  const int cb   = blockIdx.x * BNX;
  const int n0   = cb + wcol * 64;

  // A-fragment (16-bit A 16x32 layout): lanes 0-15 rows m, K chunks {0..7,16..23};
  // lanes 16-31 same rows, chunks {8..15,24..31}.
  const int khalf = (lane >> 4) << 3;         // 0 or 8
  // B-fragment: lane n = lane&15 ; same K chunks as A: {0..7,16..23} / {8..15,24..31}.
  const int nfr  = lane & 15;
  const int kfrB = (lane >> 4) << 3;          // 0 or 8

  const bf16* arow0 = A + (size_t)(m0 + (lane & 15)) * lda;
  const bf16* arow1 = arow0 + (size_t)16 * lda;
  const bf16* brow[4];
#pragma unroll
  for (int ni = 0; ni < 4; ++ni)
    brow[ni] = Bt + (size_t)(n0 + ni * 16 + nfr) * ldbt;

  v8f acc[2][4];
#pragma unroll
  for (int mi = 0; mi < 2; ++mi)
#pragma unroll
    for (int ni = 0; ni < 4; ++ni)
#pragma unroll
      for (int i = 0; i < 8; ++i) acc[mi][ni][i] = 0.0f;

  const bool full = (cb + BNX <= Ncols) && ((Kdim & (BKX - 1)) == 0);

  if (full) {
    for (int kb = 0; kb < Kdim; kb += BKX) {
      AF af[2], bfr[4];
      af[0].h[0] = *(const v8bf*)(arow0 + kb + khalf);
      af[0].h[1] = *(const v8bf*)(arow0 + kb + 16 + khalf);
      af[1].h[0] = *(const v8bf*)(arow1 + kb + khalf);
      af[1].h[1] = *(const v8bf*)(arow1 + kb + 16 + khalf);
#pragma unroll
      for (int ni = 0; ni < 4; ++ni) {
        bfr[ni].h[0] = *(const v8bf*)(brow[ni] + kb + kfrB);
        bfr[ni].h[1] = *(const v8bf*)(brow[ni] + kb + 16 + kfrB);
      }
      if (kb + BKX < Kdim) {                  // gfx1250 global_prefetch of next weight K-tile
#pragma unroll
        for (int ni = 0; ni < 4; ++ni)
          __builtin_prefetch(brow[ni] + kb + BKX, 0, 3);
      }
#pragma unroll
      for (int mi = 0; mi < 2; ++mi)
#pragma unroll
        for (int ni = 0; ni < 4; ++ni)
          acc[mi][ni] = __builtin_amdgcn_wmma_f32_16x16x32_bf16(
              false, af[mi].v, false, bfr[ni].v, (short)0, acc[mi][ni], false, false);
    }
  } else {
    for (int kb = 0; kb < Kdim; kb += BKX) {
      AF af[2], bfr[4];
      af[0].h[0] = ld8(arow0 + kb + khalf,      kb + khalf + 8       <= Kdim);
      af[0].h[1] = ld8(arow0 + kb + 16 + khalf, kb + 16 + khalf + 8  <= Kdim);
      af[1].h[0] = ld8(arow1 + kb + khalf,      kb + khalf + 8       <= Kdim);
      af[1].h[1] = ld8(arow1 + kb + 16 + khalf, kb + 16 + khalf + 8  <= Kdim);
#pragma unroll
      for (int ni = 0; ni < 4; ++ni) {
        bool okc = (n0 + ni * 16 + nfr) < Ncols;
        bfr[ni].h[0] = ld8(brow[ni] + kb + kfrB,     okc && (kb + kfrB + 8  <= Kdim));
        bfr[ni].h[1] = ld8(brow[ni] + kb + 16 + kfrB, okc && (kb + 16 + kfrB + 8 <= Kdim));
      }
#pragma unroll
      for (int mi = 0; mi < 2; ++mi)
#pragma unroll
        for (int ni = 0; ni < 4; ++ni)
          acc[mi][ni] = __builtin_amdgcn_wmma_f32_16x16x32_bf16(
              false, af[mi].v, false, bfr[ni].v, (short)0, acc[mi][ni], false, false);
    }
  }

  // ---- store with fused epilogue ----
  const int rb = (lane >> 4) << 3;            // lanes>=16 hold rows +8..+15
#pragma unroll
  for (int mi = 0; mi < 2; ++mi) {
    const int rowb = m0 + mi * 16 + rb;
#pragma unroll
    for (int ni = 0; ni < 4; ++ni) {
      const int col = n0 + ni * 16 + nfr;
      if (full || col < Ncols) {
#pragma unroll
        for (int r = 0; r < 8; ++r) {
          size_t idx = (size_t)(rowb + r) * ldc + col;
          float v = acc[mi][ni][r];
          if (EPI == 1) v += aux[idx];
          if (EPI == 2) {
            v += aux[col];
            v = (v > 20.0f) ? v : log1pf(__expf(v));   // softplus
          }
          C[idx] = v;
        }
      }
    }
  }
}

// ---------------- elementwise / utility kernels ----------------
__global__ void zero_f32(float* p, int n) {
  int i = blockIdx.x * 256 + threadIdx.x;
  if (i < n) p[i] = 0.0f;
}

__global__ void cast_f32_bf16(const float* __restrict__ in, bf16* __restrict__ out, int n) {
  int i = blockIdx.x * 256 + threadIdx.x;
  if (i < n) out[i] = (bf16)in[i];
}

// per-layer transpose + cast: in (layer, K, N) f32 -> out (layer, N, K) bf16
__global__ void cast_transpose_bf16(const float* __restrict__ in, bf16* __restrict__ out,
                                    int K, int N) {
  size_t lo = (size_t)blockIdx.y * K * N;
  int idx = blockIdx.x * 256 + threadIdx.x;
  if (idx < K * N) {
    int n = idx / K, k = idx - n * K;
    out[lo + (size_t)n * K + k] = (bf16)in[lo + (size_t)k * N + n];
  }
}

__global__ void embed_gather(const int* __restrict__ ids, const float* __restrict__ embed,
                             float* __restrict__ x) {
  int t  = blockIdx.x;
  int id = ids[t];
  for (int c = threadIdx.x; c < D_; c += blockDim.x)
    x[(size_t)t * D_ + c] = embed[(size_t)id * D_ + c];
}

__global__ __launch_bounds__(256) void rmsnorm_to_bf16(const float* __restrict__ x,
                                                       const float* __restrict__ w,
                                                       bf16* __restrict__ out) {
  __shared__ float red[256];
  int t = blockIdx.x;
  const float* xr = x + (size_t)t * D_;
  float s = 0.0f;
  for (int c = threadIdx.x; c < D_; c += 256) { float v = xr[c]; s += v * v; }
  red[threadIdx.x] = s;
  __syncthreads();
  for (int o = 128; o > 0; o >>= 1) {
    if (threadIdx.x < o) red[threadIdx.x] += red[threadIdx.x + o];
    __syncthreads();
  }
  float scale = rsqrtf(red[0] / (float)D_ + EPS_);
  for (int c = threadIdx.x; c < D_; c += 256)
    out[(size_t)t * D_ + c] = (bf16)(xr[c] * scale * w[c]);
}

// depthwise causal conv (K=4) + bias + silu + mask; reads u-half of uz
__global__ __launch_bounds__(256) void conv_silu(const float* __restrict__ uz,
                                                 const float* __restrict__ cw,
                                                 const float* __restrict__ cbi,
                                                 const int* __restrict__ mask,
                                                 float* __restrict__ ucs,
                                                 bf16* __restrict__ ubf) {
  int idx = blockIdx.x * 256 + threadIdx.x;   // t*DI + d
  int t = idx / DI_, d = idx % DI_;
  int b = t / L_,  l = t % L_;
  float acc = 0.0f;
#pragma unroll
  for (int j = 0; j < KC_; ++j) {
    int lj = l - (KC_ - 1) + j;
    if (lj >= 0) {
      int tj = b * L_ + lj;
      acc += cw[d * KC_ + j] * uz[(size_t)tj * (2 * DI_) + d] * (float)mask[tj];
    }
  }
  acc += cbi[d];
  float sv = siluf(acc) * (float)mask[t];
  ucs[idx] = sv;
  ubf[idx] = (bf16)sv;
}

// sequential SSM scan, one thread per (b, d) channel; 16 states in registers.
// The whole B/C slab for this block's batch (256 tokens x 32 floats, stride-80 rows)
// is staged into LDS up-front by the Tensor Data Mover (one tensor_load_to_lds issued
// by wave 0, tracked by TENSORcnt), so the 256-step recurrence runs with zero
// per-step barriers or global staging loads on its critical path.
__global__ __launch_bounds__(256) void ssm_scan(const float* __restrict__ dt,
                                                const float* __restrict__ ucs,
                                                const float* __restrict__ xdb,
                                                const float* __restrict__ A_log,
                                                const float* __restrict__ Dp,
                                                float* __restrict__ states,
                                                float* __restrict__ y) {
  __shared__ __attribute__((aligned(16))) float sBC[L_][2 * N_];   // 32 KB
  const int bpb = DI_ / 256;                   // blocks per batch = 6
  const int b   = blockIdx.x / bpb;            // uniform per block (SGPR)
  const int d   = (blockIdx.x % bpb) * 256 + threadIdx.x;

  if (threadIdx.x < 32) {
    // Build the TDM descriptor (D#) in SGPRs: groups 0 (4 dw) and 1 (8 dw).
    const float* gsrc = xdb + (size_t)(b * L_) * XDBW + R_;   // tile start (col 48)
    unsigned long long ga = (unsigned long long)(size_t)gsrc; // 57-bit global addr
    unsigned lds_off = (unsigned)(size_t)(void*)&sBC[0][0];   // LDS byte offset (low 32b)
    u32x4 g0;
    g0[0] = 1u;                                    // count=1, user descriptor
    g0[1] = lds_off;                               // lds_addr
    g0[2] = (unsigned)ga;                          // global_addr[31:0]
    g0[3] = (unsigned)(ga >> 32) | (2u << 30);     // global_addr[56:32] | type=2
    u32x8 g1;
    g1[0] = 2u << 16;                              // data_size = 4 bytes
    g1[1] = ((unsigned)(2 * N_)) << 16;            // tensor_dim0 = 32 (bits 79:48 lo)
    g1[2] = ((unsigned)L_) << 16;                  // tensor_dim1 = 256 (bits 111:80 lo)
    g1[3] = ((unsigned)(2 * N_)) << 16;            // tile_dim0 = 32 (bits 127:112)
    g1[4] = (unsigned)L_;                          // tile_dim1 = 256 (bits 143:128)
    g1[5] = (unsigned)XDBW;                        // tensor_dim0_stride = 80 (bits 207:160)
    g1[6] = 0u;
    g1[7] = 0u;
    asm volatile("tensor_load_to_lds %0, %1" :: "s"(g0), "s"(g1) : "memory");
    __builtin_amdgcn_s_wait_tensorcnt(0);
  }
  __syncthreads();

  float h[N_], An[N_];
#pragma unroll
  for (int n = 0; n < N_; ++n) {
    h[n]  = states[((size_t)b * DI_ + d) * N_ + n];
    An[n] = -__expf(A_log[(size_t)d * N_ + n]);
  }
  float dpv = Dp[d];
  for (int l = 0; l < L_; ++l) {
    int t = b * L_ + l;
    float dtv = dt[(size_t)t * DI_ + d];
    float uv  = ucs[(size_t)t * DI_ + d];
    float du  = dtv * uv;
    float yv  = 0.0f;
#pragma unroll
    for (int n = 0; n < N_; ++n) {
      h[n] = __expf(dtv * An[n]) * h[n] + du * sBC[l][n];
      yv += h[n] * sBC[l][N_ + n];
    }
    y[(size_t)t * DI_ + d] = yv + uv * dpv;
  }
#pragma unroll
  for (int n = 0; n < N_; ++n) states[((size_t)b * DI_ + d) * N_ + n] = h[n];
}

__global__ void gate_silu(const float* __restrict__ y, const float* __restrict__ uz,
                          bf16* __restrict__ out) {
  int idx = blockIdx.x * 256 + threadIdx.x;   // t*DI + d
  int t = idx / DI_, d = idx % DI_;
  float z = uz[(size_t)t * (2 * DI_) + DI_ + d];
  out[idx] = (bf16)(y[idx] * siluf(z));
}

// ---------------- host orchestration ----------------
extern "C" void kernel_launch(void* const* d_in, const int* in_sizes, int n_in,
                              void* d_out, int out_size, void* d_ws, size_t ws_size,
                              hipStream_t stream) {
  (void)in_sizes; (void)n_in; (void)out_size; (void)ws_size;
  const int*   q_ids   = (const int*)d_in[0];
  const int*   q_mask  = (const int*)d_in[1];
  const int*   a_ids   = (const int*)d_in[2];
  const int*   a_mask  = (const int*)d_in[3];
  const float* p_embed = (const float*)d_in[4];
  const float* p_Win   = (const float*)d_in[5];
  const float* p_convw = (const float*)d_in[6];
  const float* p_convb = (const float*)d_in[7];
  const float* p_Wx    = (const float*)d_in[8];
  const float* p_Wdt   = (const float*)d_in[9];
  const float* p_bdt   = (const float*)d_in[10];
  const float* p_Alog  = (const float*)d_in[11];
  const float* p_Dp    = (const float*)d_in[12];
  const float* p_Wout  = (const float*)d_in[13];
  const float* p_lnw   = (const float*)d_in[14];
  const float* p_lnfw  = (const float*)d_in[15];

  char* ws = (char*)d_ws;
  size_t off = 0;
  auto alloc = [&](size_t bytes) -> void* {
    void* p = ws + off;
    off += (bytes + 255) & ~(size_t)255;
    return p;
  };

  // transposed bf16 weights: [layer][N][K]
  bf16*  embed_bf = (bf16*)alloc((size_t)V_ * D_ * 2);               // already [V][D] = [N][K]
  bf16*  Win_bt   = (bf16*)alloc((size_t)NL_ * 2 * DI_ * D_ * 2);    // [2DI][768]
  bf16*  Wx_bt    = (bf16*)alloc((size_t)NL_ * XDBW * DI_ * 2);      // [80][1536]
  bf16*  Wdt_bt   = (bf16*)alloc((size_t)NL_ * DI_ * R_ * 2);        // [1536][48]
  bf16*  Wout_bt  = (bf16*)alloc((size_t)NL_ * D_ * DI_ * 2);        // [768][1536]
  float* x        = (float*)alloc((size_t)T_ * D_ * 4);
  bf16*  xn       = (bf16*)alloc((size_t)T_ * D_ * 2);
  float* uz       = (float*)alloc((size_t)T_ * 2 * DI_ * 4);
  float* ucs      = (float*)alloc((size_t)T_ * DI_ * 4);
  bf16*  ubf      = (bf16*)alloc((size_t)T_ * DI_ * 2);
  float* xdb      = (float*)alloc((size_t)T_ * XDBW * 4);
  bf16*  xdbb     = (bf16*)alloc((size_t)T_ * XDBW * 2);
  float* dtb      = (float*)alloc((size_t)T_ * DI_ * 4);
  float* yb       = (float*)alloc((size_t)T_ * DI_ * 4);
  bf16*  ygb      = (bf16*)alloc((size_t)T_ * DI_ * 2);
  float* states   = (float*)alloc((size_t)NL_ * B_ * DI_ * N_ * 4);

  // one-time (per launch) weight downconversion fp32 -> bf16 (+ transpose to [N][K])
  {
    int n;
    n = V_ * D_;
    cast_f32_bf16<<<(n + 255) / 256, 256, 0, stream>>>(p_embed, embed_bf, n);
    n = D_ * 2 * DI_;
    cast_transpose_bf16<<<dim3((n + 255) / 256, NL_), 256, 0, stream>>>(p_Win, Win_bt, D_, 2 * DI_);
    n = DI_ * XDBW;
    cast_transpose_bf16<<<dim3((n + 255) / 256, NL_), 256, 0, stream>>>(p_Wx, Wx_bt, DI_, XDBW);
    n = R_ * DI_;
    cast_transpose_bf16<<<dim3((n + 255) / 256, NL_), 256, 0, stream>>>(p_Wdt, Wdt_bt, R_, DI_);
    n = DI_ * D_;
    cast_transpose_bf16<<<dim3((n + 255) / 256, NL_), 256, 0, stream>>>(p_Wout, Wout_bt, DI_, D_);
    n = NL_ * B_ * DI_ * N_;
    zero_f32<<<(n + 255) / 256, 256, 0, stream>>>(states, n);
  }

  for (int run = 0; run < 2; ++run) {
    const int* ids  = run ? a_ids  : q_ids;
    const int* mask = run ? a_mask : q_mask;

    embed_gather<<<T_, 256, 0, stream>>>(ids, p_embed, x);

    for (int layer = 0; layer < NL_; ++layer) {
      rmsnorm_to_bf16<<<T_, 256, 0, stream>>>(x, p_lnw + (size_t)layer * D_, xn);

      // uz = xn @ W_in  (512 x 3072 x 768)
      gemm_bf16_wmma<0><<<dim3((2 * DI_) / BNX, T_ / BMX), 256, 0, stream>>>(
          xn, Win_bt + (size_t)layer * 2 * DI_ * D_, uz, nullptr,
          2 * DI_, D_, D_, D_, 2 * DI_);

      conv_silu<<<(T_ * DI_) / 256, 256, 0, stream>>>(
          uz, p_convw + (size_t)layer * DI_ * KC_, p_convb + (size_t)layer * DI_,
          mask, ucs, ubf);

      // xdb = u @ W_x   (512 x 80 x 1536)
      gemm_bf16_wmma<0><<<dim3((XDBW + BNX - 1) / BNX, T_ / BMX), 256, 0, stream>>>(
          ubf, Wx_bt + (size_t)layer * XDBW * DI_, xdb, nullptr,
          XDBW, DI_, DI_, DI_, XDBW);

      cast_f32_bf16<<<(T_ * XDBW + 255) / 256, 256, 0, stream>>>(xdb, xdbb, T_ * XDBW);

      // dt = softplus(xdb[:, :R] @ W_dt + b_dt)   (512 x 1536 x 48)
      gemm_bf16_wmma<2><<<dim3(DI_ / BNX, T_ / BMX), 256, 0, stream>>>(
          xdbb, Wdt_bt + (size_t)layer * DI_ * R_, dtb, p_bdt + (size_t)layer * DI_,
          DI_, R_, XDBW, R_, DI_);

      ssm_scan<<<(B_ * DI_) / 256, 256, 0, stream>>>(
          dtb, ucs, xdb, p_Alog + (size_t)layer * DI_ * N_, p_Dp + (size_t)layer * DI_,
          states + (size_t)layer * B_ * DI_ * N_, yb);

      gate_silu<<<(T_ * DI_) / 256, 256, 0, stream>>>(yb, uz, ygb);

      // x += yg @ W_out   (512 x 768 x 1536), residual fused in epilogue
      gemm_bf16_wmma<1><<<dim3(D_ / BNX, T_ / BMX), 256, 0, stream>>>(
          ygb, Wout_bt + (size_t)layer * D_ * DI_, x, x,
          D_, DI_, DI_, DI_, D_);
    }

    if (run == 1) {
      rmsnorm_to_bf16<<<T_, 256, 0, stream>>>(x, p_lnfw, xn);
      // logits = xn @ embed^T   (512 x 50280 x 768); embed is already [V][D] = Bt
      gemm_bf16_wmma<0><<<dim3((V_ + BNX - 1) / BNX, T_ / BMX), 256, 0, stream>>>(
          xn, embed_bf, (float*)d_out, nullptr,
          V_, D_, D_, D_, V_);
    }
  }
}
